// BEVFormerEncoder_10170482557432
// MI455X (gfx1250) — matrix operand order repaired and checked by reading, not compiled, not run on hardware
//
#include <hip/hip_runtime.h>
#include <hip/hip_bf16.h>

// ---------------------------------------------------------------------------
// BEVFormer encoder for gfx1250 (MI455X).
// Dense projections: v_wmma_f32_16x16x32_bf16 (bf16 A/B, f32 accum),
//   A tile staged in LDS (global_load_async_to_lds_b128 when available),
//   shared by 8 waves per block.
// Deformable sampling: VALU gather kernels (L2-resident working set).
// LayerNorm: wave32 shuffle reduction.
// ---------------------------------------------------------------------------

#define NQv     10000
#define Cv      256
#define Hv      8
#define HDv     32
#define Lv      4
#define Pv      4
#define Dv      4
#define CAMSv   6
#define FFNv    1024
#define LAYERSv 3
#define BEVH    100
#define BEVW    100
#define STOT    19560

typedef __attribute__((ext_vector_type(16))) __bf16 v16bf;
typedef __attribute__((ext_vector_type(8)))  float  v8f;

#if __has_builtin(__builtin_amdgcn_global_load_async_to_lds_b128) && \
    __has_builtin(__builtin_amdgcn_s_wait_asynccnt)
#define USE_ASYNC_LDS 1
typedef int v4i_ __attribute__((__vector_size__(4 * sizeof(int))));
typedef __attribute__((address_space(1))) v4i_ gas_v4i;
typedef __attribute__((address_space(3))) v4i_ las_v4i;
#else
#define USE_ASYNC_LDS 0
#endif

__device__ __constant__ int c_lvl_h[4] = {92, 46, 23, 12};
__device__ __constant__ int c_lvl_w[4] = {160, 80, 40, 20};
__device__ __constant__ int c_lvl_s[4] = {0, 14720, 18400, 19320};

// ---------------------------- elementwise utils ----------------------------

__global__ __launch_bounds__(256) void k_copy(float* __restrict__ dst,
                                              const float* __restrict__ src, int n) {
  int i = blockIdx.x * 256 + threadIdx.x;
  if (i < n) dst[i] = src[i];
}

__global__ __launch_bounds__(256) void k_add(float* __restrict__ out,
                                             const float* __restrict__ a,
                                             const float* __restrict__ b, int n) {
  int i = blockIdx.x * 256 + threadIdx.x;
  if (i < n) out[i] = a[i] + b[i];
}

// Wt[n*K + k] = bf16(W[k*N + n])  (N-major so B fragments are contiguous)
__global__ __launch_bounds__(256) void k_wt_bf16_t(const float* __restrict__ W,
                                                   __bf16* __restrict__ Wt,
                                                   int K, int N) {
  int i = blockIdx.x * 256 + threadIdx.x;
  if (i >= K * N) return;
  int n = i / K, k = i - n * K;
  Wt[i] = (__bf16)W[k * N + n];
}

// ------------------------------- WMMA GEMM ---------------------------------
// Out[M,N] = A[M,K](f32->bf16) * Wt[N,K](bf16) + bias (+resid) (+relu)
// 256 thr = 8 waves per block; block tile 16 x 128 (wave w owns cols w*16..).
// Per k-step: A 16x32 f32 tile staged in LDS (async-to-LDS if available),
// shared by all 8 waves; B fragment = contiguous 32B of pre-transposed bf16.

__global__ __launch_bounds__(256) void k_gemm(const float* __restrict__ A,
                                              const __bf16* __restrict__ Wt,
                                              const float* __restrict__ bias,
                                              const float* __restrict__ resid,
                                              float* __restrict__ Out,
                                              int M, int N, int K, int relu) {
  __shared__ float lds_a[16 * 32];            // 2 KB A tile (f32)

  const int t    = threadIdx.x;
  const int lane = t & 31;
  const int wave = t >> 5;
  const int tm = blockIdx.x;
  const int tn = blockIdx.y * 8 + wave;
  const bool active = (tn * 16) < N;          // wave-uniform

  const int hi = lane >> 4;                   // 0: lanes 0-15, 1: lanes 16-31
  const int ln = lane & 15;
  const int ncol = active ? tn * 16 + ln : 0;
  const __bf16* brow = Wt + (size_t)ncol * K + hi * 16;

  // staging assignment: threads 0..127 each move 4 consecutive floats (16 B)
  int s_m = 0, s_k = 0;
  const float* s_src_row = A;
  if (t < 128) {
    int flat = t * 4;                         // 512 floats = 16x32 tile
    s_m = flat >> 5;                          // row in tile
    s_k = flat & 31;                          // col in tile
    int gm = tm * 16 + s_m;
    if (gm >= M) gm = M - 1;                  // clamped load; store guarded
    s_src_row = A + (size_t)gm * K;
  }

  v8f acc = {};
  for (int k0 = 0; k0 < K; k0 += 32) {
    if (t < 128) {
      const float* src = s_src_row + k0 + s_k;
      float* dst = &lds_a[s_m * 32 + s_k];
#if USE_ASYNC_LDS
      __builtin_amdgcn_global_load_async_to_lds_b128(
          (gas_v4i*)src, (las_v4i*)dst, 0, 0);
#else
      float4 v4 = *(const float4*)src;
      *(float4*)dst = v4;
#endif
    }
#if USE_ASYNC_LDS
    __builtin_amdgcn_s_wait_asynccnt(0);
#endif
    __syncthreads();

    // A 16x32 bf16 fragment: k = hi*16 + e (matches B order)
    v16bf af;
    const float* arow = &lds_a[ln * 32];
#pragma unroll
    for (int e = 0; e < 8; ++e) {
      af[e]     = (__bf16)arow[hi * 16 + e];
      af[8 + e] = (__bf16)arow[hi * 16 + 8 + e];
    }
    // B 32x16 bf16 fragment: lane=col, k = hi*16 + e (contiguous in Wt)
    v16bf bfv = {};
    if (active) bfv = *(const v16bf*)(brow + k0);

    if (active) {                              // wave-uniform: EXEC all-ones
      acc = __builtin_amdgcn_wmma_f32_16x16x32_bf16(
          false, af, false, bfv, (short)0, acc, false, false);
    }
    __syncthreads();
  }

  if (active) {
    const float bv = bias ? bias[ncol] : 0.0f;
#pragma unroll
    for (int r = 0; r < 8; ++r) {
      int row = tm * 16 + hi * 8 + r;         // C/D layout: VGPR r -> M = hi*8+r
      if (row < M) {
        float v = acc[r] + bv;
        if (resid) v += resid[(size_t)row * N + ncol];
        if (relu) v = v > 0.0f ? v : 0.0f;
        Out[(size_t)row * N + ncol] = v;
      }
    }
  }
}

// ------------------------------ softmax (n<=16) ----------------------------

__global__ __launch_bounds__(256) void k_softmax(float* __restrict__ x,
                                                 int rows, int n) {
  int r = blockIdx.x * 256 + threadIdx.x;
  if (r >= rows) return;
  float* p = x + (size_t)r * n;
  float mx = -1e30f;
  for (int i = 0; i < n; ++i) mx = fmaxf(mx, p[i]);
  float s = 0.0f;
  for (int i = 0; i < n; ++i) { float e = __expf(p[i] - mx); p[i] = e; s += e; }
  float inv = 1.0f / s;
  for (int i = 0; i < n; ++i) p[i] *= inv;
}

// ------------------------- point sampling / masks --------------------------

__global__ __launch_bounds__(256) void k_point_sampling(const float* __restrict__ l2i,
                                                        float* __restrict__ rpc,
                                                        float* __restrict__ mask_any) {
  int i = blockIdx.x * 256 + threadIdx.x;     // i = c*NQ + q
  if (i >= CAMSv * NQv) return;
  int c = i / NQv, q = i - c * NQv;
  const float* Mm = l2i + c * 16;
  float rx = ((q % BEVW) + 0.5f) / (float)BEVW;
  float ry = ((q / BEVW) + 0.5f) / (float)BEVH;
  float px = rx * 102.4f - 51.2f;
  float py = ry * 102.4f - 51.2f;
  float any = 0.0f;
#pragma unroll
  for (int d = 0; d < Dv; ++d) {
    float pz = ((d + 0.5f) / (float)Dv) * 8.0f - 5.0f;
    float cx = Mm[0] * px + Mm[1] * py + Mm[2]  * pz + Mm[3];
    float cy = Mm[4] * px + Mm[5] * py + Mm[6]  * pz + Mm[7];
    float cz = Mm[8] * px + Mm[9] * py + Mm[10] * pz + Mm[11];
    bool m = cz > 1e-5f;
    float zc = fmaxf(cz, 1e-5f);
    float x = cx / zc * (1.0f / 1280.0f);
    float y = cy / zc * (1.0f / 736.0f);
    m = m && (x > 0.0f) && (x < 1.0f) && (y > 0.0f) && (y < 1.0f);
    float fm = m ? 1.0f : 0.0f;
    rpc[(size_t)i * 8 + d * 2 + 0] = fm * x;
    rpc[(size_t)i * 8 + d * 2 + 1] = fm * y;
    any = fmaxf(any, fm);
  }
  mask_any[i] = any;
}

__global__ __launch_bounds__(256) void k_count(const float* __restrict__ mask_any,
                                               float* __restrict__ cnt) {
  int q = blockIdx.x * 256 + threadIdx.x;
  if (q >= NQv) return;
  float s = 0.0f;
  for (int c = 0; c < CAMSv; ++c) s += mask_any[c * NQv + q];
  cnt[q] = fmaxf(s, 1.0f);
}

// ------------------------ temporal deformable attn -------------------------
// thread = one output element (q,h,e); single level 100x100, P=4 points.

__global__ __launch_bounds__(256) void k_temporal_attn(const float* __restrict__ v,
                                                       const float* __restrict__ off,
                                                       const float* __restrict__ aw,
                                                       float* __restrict__ out) {
  int i = blockIdx.x * 256 + threadIdx.x;
  if (i >= NQv * Cv) return;
  int q = i >> 8, he = i & 255, h = he >> 5, e = he & 31;
  float rx = ((q % BEVW) + 0.5f) / (float)BEVW;
  float ry = ((q / BEVW) + 0.5f) / (float)BEVH;
  float acc = 0.0f;
#pragma unroll
  for (int p = 0; p < Pv; ++p) {
    float lx = rx + off[(size_t)q * 64 + h * 8 + p * 2 + 0] * (1.0f / BEVW);
    float ly = ry + off[(size_t)q * 64 + h * 8 + p * 2 + 1] * (1.0f / BEVH);
    float w  = aw[(size_t)q * 32 + h * 4 + p];
    float x = lx * BEVW - 0.5f, y = ly * BEVH - 0.5f;
    float x0f = floorf(x), y0f = floorf(y);
    int x0 = (int)x0f, y0 = (int)y0f;
    float fx = x - x0f, fy = y - y0f;
#pragma unroll
    for (int dy = 0; dy < 2; ++dy)
#pragma unroll
      for (int dx = 0; dx < 2; ++dx) {
        int xi = x0 + dx, yi = y0 + dy;
        float wgt = (dx ? fx : 1.0f - fx) * (dy ? fy : 1.0f - fy);
        if (xi >= 0 && xi < BEVW && yi >= 0 && yi < BEVH)
          acc += w * wgt * v[((size_t)(yi * BEVW + xi)) * Cv + h * HDv + e];
      }
  }
  out[i] = acc;
}

// ------------------------- spatial deformable attn -------------------------
// one camera per launch; accumulates masked result into slots.

__global__ __launch_bounds__(256) void k_spatial_attn(const float* __restrict__ vc,
                                                      const float* __restrict__ off,
                                                      const float* __restrict__ aw,
                                                      const float* __restrict__ rpc_c,
                                                      const float* __restrict__ mask_c,
                                                      float* __restrict__ slots,
                                                      int first) {
  int i = blockIdx.x * 256 + threadIdx.x;
  if (i >= NQv * Cv) return;
  int q = i >> 8, h = (i >> 5) & 7, e = i & 31;
  float msk = mask_c[q];
  float acc = 0.0f;
  if (msk != 0.0f) {
#pragma unroll
    for (int l = 0; l < Lv; ++l) {
      int Hl = c_lvl_h[l], Wl = c_lvl_w[l], st = c_lvl_s[l];
      float iW = 1.0f / (float)Wl, iH = 1.0f / (float)Hl;
#pragma unroll
      for (int p = 0; p < Pv; ++p) {   // p == d (P//D == 1)
        float lx = rpc_c[(size_t)q * 8 + p * 2 + 0] +
                   off[(size_t)q * 256 + h * 32 + l * 8 + p * 2 + 0] * iW;
        float ly = rpc_c[(size_t)q * 8 + p * 2 + 1] +
                   off[(size_t)q * 256 + h * 32 + l * 8 + p * 2 + 1] * iH;
        float w = aw[(size_t)q * 128 + h * 16 + l * 4 + p];
        float x = lx * Wl - 0.5f, y = ly * Hl - 0.5f;
        float x0f = floorf(x), y0f = floorf(y);
        int x0 = (int)x0f, y0 = (int)y0f;
        float fx = x - x0f, fy = y - y0f;
#pragma unroll
        for (int dy = 0; dy < 2; ++dy)
#pragma unroll
          for (int dx = 0; dx < 2; ++dx) {
            int xi = x0 + dx, yi = y0 + dy;
            float wgt = (dx ? fx : 1.0f - fx) * (dy ? fy : 1.0f - fy);
            if (xi >= 0 && xi < Wl && yi >= 0 && yi < Hl)
              acc += w * wgt * vc[((size_t)(st + yi * Wl + xi)) * Cv + h * 32 + e];
          }
      }
    }
  }
  if (first) slots[i] = acc;
  else       slots[i] += acc;
}

__global__ __launch_bounds__(256) void k_divcount(float* __restrict__ x,
                                                  const float* __restrict__ cnt) {
  int i = blockIdx.x * 256 + threadIdx.x;
  if (i >= NQv * Cv) return;
  x[i] /= cnt[i >> 8];
}

// ------------------------------- LayerNorm ---------------------------------
// one wave32 per row of 256; 8 elems per lane; shuffle reduction.

__global__ __launch_bounds__(256) void k_layernorm(float* __restrict__ out,
                                                   const float* __restrict__ x,
                                                   const float* __restrict__ g,
                                                   const float* __restrict__ b) {
  int lane = threadIdx.x & 31;
  int row = blockIdx.x * 8 + (threadIdx.x >> 5);
  if (row >= NQv) return;
  const float* xr = x + (size_t)row * Cv;
  float vals[8];
  float s = 0.0f;
#pragma unroll
  for (int j = 0; j < 8; ++j) { vals[j] = xr[lane + 32 * j]; s += vals[j]; }
#pragma unroll
  for (int o = 16; o > 0; o >>= 1) s += __shfl_xor(s, o, 32);
  float mean = s * (1.0f / Cv);
  float var = 0.0f;
#pragma unroll
  for (int j = 0; j < 8; ++j) { float d = vals[j] - mean; var += d * d; }
#pragma unroll
  for (int o = 16; o > 0; o >>= 1) var += __shfl_xor(var, o, 32);
  float inv = rsqrtf(var * (1.0f / Cv) + 1e-5f);
#pragma unroll
  for (int j = 0; j < 8; ++j) {
    int c2 = lane + 32 * j;
    out[(size_t)row * Cv + c2] = (vals[j] - mean) * inv * g[c2] + b[c2];
  }
}

// ------------------------------- host side ---------------------------------

// param insertion-order indices within a layer's dict (26 tensors/layer)
enum {
  I_TOFF_W = 0, I_TOFF_B, I_TAW_W, I_TAW_B, I_TV_W, I_TV_B, I_TO_W, I_TO_B,
  I_SOFF_W, I_SOFF_B, I_SAW_W, I_SAW_B, I_SV_W, I_SV_B, I_SO_W, I_SO_B,
  I_F1_W, I_F1_B, I_F2_W, I_F2_B,
  I_N1_G, I_N1_B, I_N2_G, I_N2_B, I_N3_G, I_N3_B
};

static inline int ceil_div(int a, int b) { return (a + b - 1) / b; }

extern "C" void kernel_launch(void* const* d_in, const int* in_sizes, int n_in,
                              void* d_out, int out_size, void* d_ws, size_t ws_size,
                              hipStream_t stream) {
  const float* bev_query = (const float*)d_in[0];
  const float* key       = (const float*)d_in[1];
  const float* bev_pos   = (const float*)d_in[2];
  const float* prev_bev  = (const float*)d_in[3];
  const float* lidar2img = (const float*)d_in[4];
  auto PP = [&](int layer, int idx) -> const float* {
    return (const float*)d_in[5 + layer * 26 + idx];
  };

  // workspace layout (floats, 256-elt aligned)
  float* ws = (float*)d_ws;
  size_t o = 0;
  auto alloc = [&](size_t nflt) { size_t r = o; o += (nflt + 255) & ~size_t(255); return r; };
  __bf16* wt   = (__bf16*)(ws + alloc(262144 / 2 + 256)); // 262144 bf16 max weight
  float* qp    = ws + alloc((size_t)NQv * Cv);
  float* tmp   = ws + alloc((size_t)NQv * Cv);
  float* slots = ws + alloc((size_t)NQv * Cv);
  float* big   = ws + alloc((size_t)NQv * FFNv);
  float* rpc   = ws + alloc((size_t)CAMSv * NQv * 8);
  float* mska  = ws + alloc((size_t)CAMSv * NQv);
  float* cnt   = ws + alloc((size_t)NQv);
  // phase-aliased views of `big`
  float* v_t   = big;                          // temporal value  (NQ*C)
  float* off_t = big + (size_t)NQv * Cv;       // NQ*64
  float* aw_t  = off_t + (size_t)NQv * 64;     // NQ*32
  float* vc    = big;                          // spatial value   (STOT*C)
  float* off_s = big + (size_t)STOT * Cv;      // NQ*256
  float* aw_s  = off_s + (size_t)NQv * 256;    // NQ*128
  float* ff1   = big;                          // FFN hidden      (NQ*1024)

  float* q = (float*)d_out;                    // q lives in d_out throughout

  const int NC = NQv * Cv;
  const int EB = 256;

  auto gemm = [&](const float* A, const float* W, const float* bias,
                  const float* resid, float* Out, int M, int N, int K, int relu) {
    k_wt_bf16_t<<<ceil_div(K * N, EB), EB, 0, stream>>>(W, wt, K, N);
    dim3 g(ceil_div(M, 16), ceil_div(N, 128));
    k_gemm<<<g, 256, 0, stream>>>(A, wt, bias, resid, Out, M, N, K, relu);
  };

  // q = bev_query ; precompute reference-point projections (layer-invariant)
  k_copy<<<ceil_div(NC, EB), EB, 0, stream>>>(q, bev_query, NC);
  k_point_sampling<<<ceil_div(CAMSv * NQv, EB), EB, 0, stream>>>(lidar2img, rpc, mska);
  k_count<<<ceil_div(NQv, EB), EB, 0, stream>>>(mska, cnt);

  for (int layer = 0; layer < LAYERSv; ++layer) {
    // ---- temporal self-attention ----
    k_add<<<ceil_div(NC, EB), EB, 0, stream>>>(qp, q, bev_pos, NC);
    gemm(prev_bev, PP(layer, I_TV_W), PP(layer, I_TV_B), nullptr, v_t, NQv, Cv, Cv, 0);
    gemm(qp, PP(layer, I_TOFF_W), PP(layer, I_TOFF_B), nullptr, off_t, NQv, 64, Cv, 0);
    gemm(qp, PP(layer, I_TAW_W), PP(layer, I_TAW_B), nullptr, aw_t, NQv, 32, Cv, 0);
    k_softmax<<<ceil_div(NQv * Hv, EB), EB, 0, stream>>>(aw_t, NQv * Hv, 4);
    k_temporal_attn<<<ceil_div(NC, EB), EB, 0, stream>>>(v_t, off_t, aw_t, slots);
    gemm(slots, PP(layer, I_TO_W), PP(layer, I_TO_B), q, tmp, NQv, Cv, Cv, 0);
    k_layernorm<<<ceil_div(NQv, 8), 256, 0, stream>>>(q, tmp, PP(layer, I_N1_G), PP(layer, I_N1_B));

    // ---- spatial cross-attention ----
    k_add<<<ceil_div(NC, EB), EB, 0, stream>>>(qp, q, bev_pos, NC);
    gemm(qp, PP(layer, I_SOFF_W), PP(layer, I_SOFF_B), nullptr, off_s, NQv, 256, Cv, 0);
    gemm(qp, PP(layer, I_SAW_W), PP(layer, I_SAW_B), nullptr, aw_s, NQv, 128, Cv, 0);
    k_softmax<<<ceil_div(NQv * Hv, EB), EB, 0, stream>>>(aw_s, NQv * Hv, 16);
    for (int cam = 0; cam < CAMSv; ++cam) {
      gemm(key + (size_t)cam * STOT * Cv, PP(layer, I_SV_W), PP(layer, I_SV_B),
           nullptr, vc, STOT, Cv, Cv, 0);
      k_spatial_attn<<<ceil_div(NC, EB), EB, 0, stream>>>(
          vc, off_s, aw_s, rpc + (size_t)cam * NQv * 8, mska + (size_t)cam * NQv,
          slots, cam == 0 ? 1 : 0);
    }
    k_divcount<<<ceil_div(NC, EB), EB, 0, stream>>>(slots, cnt);
    gemm(slots, PP(layer, I_SO_W), PP(layer, I_SO_B), q, tmp, NQv, Cv, Cv, 0);
    k_layernorm<<<ceil_div(NQv, 8), 256, 0, stream>>>(q, tmp, PP(layer, I_N2_G), PP(layer, I_N2_B));

    // ---- FFN ----
    gemm(q, PP(layer, I_F1_W), PP(layer, I_F1_B), nullptr, ff1, NQv, FFNv, Cv, 1);
    gemm(ff1, PP(layer, I_F2_W), PP(layer, I_F2_B), q, tmp, NQv, Cv, FFNv, 0);
    k_layernorm<<<ceil_div(NQv, 8), 256, 0, stream>>>(q, tmp, PP(layer, I_N3_G), PP(layer, I_N3_B));
  }
}
